// Codec_46926812676249
// MI455X (gfx1250) — compile-verified, matrix operand order checked by reading, not run on hardware
//
#include <hip/hip_runtime.h>
#include <hip/hip_bf16.h>
#include <math.h>

// ---------- types ----------
typedef __attribute__((ext_vector_type(16))) __bf16         v16bf;
typedef __attribute__((ext_vector_type(8)))  float          v8f;
typedef __attribute__((ext_vector_type(8)))  unsigned short ushort8;
typedef __attribute__((ext_vector_type(16))) unsigned short ushort16;

union Frag { v16bf bf; ushort16 us; };

#define TPB      256
#define TOK_TILE 256          // tokens per workgroup (2 row-tiles of 16 per wave)
#define DPAD     136          // 128 dims + 8 halves pad: 272B row stride, 16B aligned
#define NCODE    256
#define NDIM     128
#define NLAYER   5
#define TDIM     8192
#define MTOK     131072       // B*T = 16*8192

// ---------- helpers ----------
__device__ __forceinline__ unsigned short f2bf(float f) {
    union { float f; unsigned u; } v; v.f = f;
    unsigned r = v.u + 0x7FFFu + ((v.u >> 16) & 1u);   // round-to-nearest-even
    return (unsigned short)(r >> 16);
}
__device__ __forceinline__ float bf2f(unsigned short h) {
    union { unsigned u; float f; } v; v.u = ((unsigned)h) << 16; return v.f;
}
// One 16-elem bf16 fragment: elems [0..7] = base[0..7], [8..15] = base[16..23]
// (CDNA5 16-bit A/B layout: per-lane K groups {h*8..h*8+7, 16+h*8..16+h*8+7})
__device__ __forceinline__ ushort16 ldfrag(const unsigned short* base) {
    ushort8 lo = *(const ushort8*)(base);
    ushort8 hi = *(const ushort8*)(base + 16);
    return __builtin_shufflevector(lo, hi, 0,1,2,3,4,5,6,7,8,9,10,11,12,13,14,15);
}

// ---------- kernels ----------
__global__ void rvq_zero(float* ws, int n) {
    int i = blockIdx.x * blockDim.x + threadIdx.x;
    if (i < n) ws[i] = 0.0f;
}

__global__ __launch_bounds__(TPB)
void rvq_main(const float* __restrict__ x, const float* __restrict__ emb,
              float* __restrict__ outQ, float* __restrict__ outI,
              float* __restrict__ counts, float* __restrict__ sums)
{
    extern __shared__ __align__(16) unsigned char smem[];
    unsigned short* embL = (unsigned short*)smem;              // 256*136*2 = 69632 B
    unsigned short* res  = embL + NCODE * DPAD;                // 256*136*2 = 69632 B
    unsigned short* xo   = res  + TOK_TILE * DPAD;             // 69632 B
    float* enorm         = (float*)(xo + TOK_TILE * DPAD);     // 1024 B
    int*   idxs          = (int*)(enorm + NCODE);              // 1024 B

    const int tid  = threadIdx.x;
    const int lane = tid & 31;
    const int wave = tid >> 5;
    const int hf   = lane >> 4;       // wave half (K / M-block selector)
    const int l15  = lane & 15;
    const int m0   = blockIdx.x * TOK_TILE;
    const int bb   = m0 >> 13;        // / 8192
    const int t0   = m0 & (TDIM - 1);

    // ---- load x tile (bf16) into residual + original-x LDS tiles ----
    for (int i = tid; i < TOK_TILE * NDIM; i += TPB) {
        int d = i >> 8, tok = i & 255;
        float v = x[((size_t)(bb * NDIM + d)) * TDIM + t0 + tok];
        unsigned short h = f2bf(v);
        res[tok * DPAD + d] = h;
        xo [tok * DPAD + d] = h;
    }

    for (int l = 0; l < NLAYER; ++l) {
        __syncthreads();   // previous layer's update phase done with embL/res

        // ---- codebook layer -> LDS bf16 ----
        const float* el = emb + (size_t)l * NCODE * NDIM;
        for (int i = tid; i < NCODE * NDIM; i += TPB) {
            int k = i >> 7, d = i & 127;
            embL[k * DPAD + d] = f2bf(el[i]);
        }
        __syncthreads();

        // ---- |e_k|^2 from the same bf16 values (consistent with WMMA dot) ----
        {
            float s = 0.0f;
            const unsigned short* row = embL + tid * DPAD;
            #pragma unroll 8
            for (int d = 0; d < NDIM; ++d) { float e = bf2f(row[d]); s += e * e; }
            enorm[tid] = s;
        }
        __syncthreads();

        // ---- per-wave 32x256 distance tile via WMMA bf16 (2 row-tiles / wave) ----
        const int rw0 = wave * 16;
        const int rw1 = 128 + wave * 16;
        const unsigned short* arow0 = res + (size_t)(rw0 + l15) * DPAD + hf * 8;
        const unsigned short* arow1 = res + (size_t)(rw1 + l15) * DPAD + hf * 8;
        Frag a0[4], a1[4];
        #pragma unroll
        for (int c = 0; c < 4; ++c) {
            a0[c].us = ldfrag(arow0 + c * 32);
            a1[c].us = ldfrag(arow1 + c * 32);
        }

        float minv0[8], minv1[8]; int mini0[8], mini1[8];
        #pragma unroll
        for (int v = 0; v < 8; ++v) {
            minv0[v] = 3.0e38f; mini0[v] = 0;
            minv1[v] = 3.0e38f; mini1[v] = 0;
        }

        for (int ct = 0; ct < 16; ++ct) {
            const int n = ct * 16 + l15;
            const unsigned short* brow = embL + (size_t)n * DPAD + hf * 8;
            Frag b0, b1, b2, b3;
            b0.us = ldfrag(brow +  0);
            b1.us = ldfrag(brow + 32);
            b2.us = ldfrag(brow + 64);
            b3.us = ldfrag(brow + 96);

            v8f acc0 = {0.f,0.f,0.f,0.f,0.f,0.f,0.f,0.f};
            v8f acc1 = {0.f,0.f,0.f,0.f,0.f,0.f,0.f,0.f};
            acc0 = __builtin_amdgcn_wmma_f32_16x16x32_bf16(false, a0[0].bf, false, b0.bf, (short)0, acc0, false, false);
            acc1 = __builtin_amdgcn_wmma_f32_16x16x32_bf16(false, a1[0].bf, false, b0.bf, (short)0, acc1, false, false);
            acc0 = __builtin_amdgcn_wmma_f32_16x16x32_bf16(false, a0[1].bf, false, b1.bf, (short)0, acc0, false, false);
            acc1 = __builtin_amdgcn_wmma_f32_16x16x32_bf16(false, a1[1].bf, false, b1.bf, (short)0, acc1, false, false);
            acc0 = __builtin_amdgcn_wmma_f32_16x16x32_bf16(false, a0[2].bf, false, b2.bf, (short)0, acc0, false, false);
            acc1 = __builtin_amdgcn_wmma_f32_16x16x32_bf16(false, a1[2].bf, false, b2.bf, (short)0, acc1, false, false);
            acc0 = __builtin_amdgcn_wmma_f32_16x16x32_bf16(false, a0[3].bf, false, b3.bf, (short)0, acc0, false, false);
            acc1 = __builtin_amdgcn_wmma_f32_16x16x32_bf16(false, a1[3].bf, false, b3.bf, (short)0, acc1, false, false);

            float en = enorm[n];
            #pragma unroll
            for (int v = 0; v < 8; ++v) {
                float d0 = en - 2.0f * acc0[v];
                float d1 = en - 2.0f * acc1[v];
                if (d0 < minv0[v]) { minv0[v] = d0; mini0[v] = n; }
                if (d1 < minv1[v]) { minv1[v] = d1; mini1[v] = n; }
            }
        }

        // ---- branchless argmin across the 16 lanes of each half ----
        #pragma unroll
        for (int off = 1; off < 16; off <<= 1) {
            #pragma unroll
            for (int v = 0; v < 8; ++v) {
                {
                    float ov = __shfl_xor(minv0[v], off, 32);
                    int   oi = __shfl_xor(mini0[v], off, 32);
                    int lt = (int)(ov < minv0[v]);
                    int eq = (int)(ov == minv0[v]);
                    int li = (int)(oi < mini0[v]);
                    int take = lt | (eq & li);
                    minv0[v] = take ? ov : minv0[v];
                    mini0[v] = take ? oi : mini0[v];
                }
                {
                    float ov = __shfl_xor(minv1[v], off, 32);
                    int   oi = __shfl_xor(mini1[v], off, 32);
                    int lt = (int)(ov < minv1[v]);
                    int eq = (int)(ov == minv1[v]);
                    int li = (int)(oi < mini1[v]);
                    int take = lt | (eq & li);
                    minv1[v] = take ? ov : minv1[v];
                    mini1[v] = take ? oi : mini1[v];
                }
            }
        }
        if (l15 == 0) {
            #pragma unroll
            for (int v = 0; v < 8; ++v) {
                idxs[rw0 + v + 8 * hf] = mini0[v];
                idxs[rw1 + v + 8 * hf] = mini1[v];
            }
        }
        __syncthreads();

        // ---- index outputs + code histogram ----
        {
            int k = idxs[tid];
            outI[(size_t)(m0 + tid) * NLAYER + l] = (float)k;
            atomicAdd(&counts[l * NCODE + k], 1.0f);
        }

        // ---- residual update, cumulative quant output, latent-loss partial ----
        float lsum = 0.0f;
        for (int i = tid; i < TOK_TILE * NDIM; i += TPB) {
            int d = i >> 8, tok = i & 255;
            int k = idxs[tok];
            float e  = bf2f(embL[k * DPAD + d]);
            float r  = bf2f(res[tok * DPAD + d]) - e;
            unsigned short rh = f2bf(r);
            res[tok * DPAD + d] = rh;
            float rr = bf2f(rh);
            float q  = bf2f(xo[tok * DPAD + d]) - rr;   // cumulative quant = x - residual
            outQ[(((size_t)(bb * NDIM + d)) * TDIM + (t0 + tok)) * NLAYER + l] = q;
            lsum += rr * rr;                            // (x - quant_l)^2 = residual^2
        }
        #pragma unroll
        for (int off = 16; off > 0; off >>= 1) lsum += __shfl_xor(lsum, off, 32);
        if (lane == 0) atomicAdd(&sums[l], lsum);
    }
}

__global__ void rvq_finalize(const float* __restrict__ counts, const float* __restrict__ sums,
                             float* __restrict__ outP, float* __restrict__ outL)
{
    __shared__ float red[256];
    const float eps = 1.1920929e-07f;
    const float Mf  = (float)MTOK;
    for (int l = 0; l < NLAYER; ++l) {
        float p = counts[l * NCODE + threadIdx.x] / Mf;
        red[threadIdx.x] = p * logf(p + eps);
        __syncthreads();
        for (int s = 128; s > 0; s >>= 1) {
            if (threadIdx.x < (unsigned)s) red[threadIdx.x] += red[threadIdx.x + s];
            __syncthreads();
        }
        if (threadIdx.x == 0) outP[l] = expf(-red[0]);
        __syncthreads();
    }
    if (threadIdx.x == 0) {
        float acc = 0.0f;
        for (int l = 0; l < NLAYER; ++l) acc += sums[l];
        outL[0] = acc / ((float)MTOK * (float)NDIM);
    }
}

// ---------- launcher ----------
extern "C" void kernel_launch(void* const* d_in, const int* in_sizes, int n_in,
                              void* d_out, int out_size, void* d_ws, size_t ws_size,
                              hipStream_t stream)
{
    (void)in_sizes; (void)n_in; (void)out_size; (void)ws_size;
    const float* x   = (const float*)d_in[0];   // [16, 128, 8192]
    const float* emb = (const float*)d_in[1];   // [5, 256, 128]

    float* out  = (float*)d_out;
    float* outQ = out;                                    // 16*128*8192*5 = 83886080
    float* outI = out + (size_t)83886080;                 // 16*8192*5     = 655360
    float* outP = outI + (size_t)655360;                  // 5
    float* outL = outP + 5;                               // 1

    float* ws     = (float*)d_ws;
    float* counts = ws;             // 5*256 = 1280
    float* sums   = ws + 1280;      // 5

    rvq_zero<<<(1285 + 255) / 256, 256, 0, stream>>>(ws, 1285);

    const size_t smemBytes =
        (size_t)NCODE * DPAD * 2 +            // embL   69632
        (size_t)TOK_TILE * DPAD * 2 * 2 +     // res+xo 139264
        (size_t)NCODE * 4 +                   // enorm  1024
        (size_t)TOK_TILE * 4;                 // idxs   1024   => 210944 B
    rvq_main<<<MTOK / TOK_TILE, TPB, smemBytes, stream>>>(x, emb, outQ, outI, counts, sums);

    rvq_finalize<<<1, 256, 0, stream>>>(counts, sums, outP, outL);
}
